// TransformLayer_2980707303697
// MI455X (gfx1250) — compile-verified, matrix operand order checked by reading, not run on hardware
//
#include <hip/hip_runtime.h>

typedef __attribute__((ext_vector_type(16))) _Float16 v16h;
typedef __attribute__((ext_vector_type(8)))  float    v8f;

namespace {
constexpr int   kND   = 200;
constexpr int   kB    = 16384;
constexpr float kStep = 1.0f / 199.0f;
constexpr float kEps  = 1e-6f;
}

__device__ __forceinline__ float softplus_precise(float x) {
    // log1p(exp(x)) overflow-safe
    return fmaxf(x, 0.0f) + log1pf(expf(-fabsf(x)));
}
__device__ __forceinline__ float softplus_fast(float x) {
    return fmaxf(x, 0.0f) + log1pf(__expf(-fabsf(x)));
}
__device__ __forceinline__ float sigmoid_fast(float x) {
    return __builtin_amdgcn_rcpf(1.0f + __expf(-x));
}

// One wave32 handles one (sample, axis) task: 200 standard grid evals + 1 eval
// at u, via 13 tiles of 16 points. Layer-1 (2->32) is done on the WMMA pipe
// with hi/lo f16 operand splitting; sigmoid + 32-wide dot + softplus on VALU.
__global__ __launch_bounds__(256) void copula_marginal_kernel(
    const float* __restrict__ U,
    const float* __restrict__ W1,
    const float* __restrict__ b1,
    const float* __restrict__ W2,
    const float* __restrict__ b2p,
    float* __restrict__ out)
{
    const int lane = threadIdx.x & 31;
    const int task = blockIdx.x * 8 + (threadIdx.x >> 5);
    const int axis = task >> 14;            // task / kB  (kB = 16384)
    const int b    = task & (kB - 1);

    const float u     = U[axis * kB + b];        // integration variable
    const float fx    = U[(1 - axis) * kB + b];  // held fixed
    const float bias2 = b2p[0];

    // ---- idx = searchsorted(grid, u, side='left'), grid[m] = m/199 ----
    int idx = (int)ceilf(u * 199.0f);
    idx = max(0, min(idx, 199));
    if (idx > 0 && (float)(idx - 1) * kStep >= u) idx -= 1;
    if (idx < 199 && (float)idx * kStep < u)      idx += 1;

    // ---- WMMA A operands: softplus(W1) with hi/lo f16 split over K slots ----
    // axis==0: feature order (g, fix) -> g uses W1 row 0; axis==1: row 1.
    const int gRow = axis;
    const int fRow = 1 - axis;

    v16h A0 = {};   // hidden 0..15
    v16h A1 = {};   // hidden 16..31
    if (lane < 16) {
        float ag0 = softplus_precise(W1[gRow * 32 + lane]);
        float af0 = softplus_precise(W1[fRow * 32 + lane]);
        float ag1 = softplus_precise(W1[gRow * 32 + 16 + lane]);
        float af1 = softplus_precise(W1[fRow * 32 + 16 + lane]);
        _Float16 agh0 = (_Float16)ag0, agl0 = (_Float16)(ag0 - (float)agh0);
        _Float16 afh0 = (_Float16)af0, afl0 = (_Float16)(af0 - (float)afh0);
        _Float16 agh1 = (_Float16)ag1, agl1 = (_Float16)(ag1 - (float)agh1);
        _Float16 afh1 = (_Float16)af1, afl1 = (_Float16)(af1 - (float)afh1);
        // K0..2: a_hi*g_hi + a_hi*g_lo + a_lo*g_hi ; K3..5: same for fix
        A0[0] = agh0; A0[1] = agh0; A0[2] = agl0;
        A0[3] = afh0; A0[4] = afh0; A0[5] = afl0;
        A1[0] = agh1; A1[1] = agh1; A1[2] = agl1;
        A1[3] = afh1; A1[4] = afh1; A1[5] = afl1;
    }

    // C (bias) and layer-2 weights matched to D layout:
    // lane n<16: rows r -> hidden base+r ; lane n>=16: rows r -> hidden base+8+r
    const int roff = (lane & 16) ? 8 : 0;
    v8f C0, C1, s2a, s2b;
#pragma unroll
    for (int r = 0; r < 8; ++r) {
        C0[r]  = b1[roff + r];
        C1[r]  = b1[16 + roff + r];
        s2a[r] = softplus_precise(W2[roff + r]);
        s2b[r] = softplus_precise(W2[16 + roff + r]);
    }

    // fixed feature (zeroed on upper lanes: B rows K 8..15, 24..31 live there, must be 0)
    const float fxm = (lane < 16) ? fx : 0.0f;
    const _Float16 fh = (_Float16)fxm;
    const _Float16 fl = (_Float16)(fxm - (float)fh);

    const int p = lane & 15;   // point slot within tile (duplicated on halves)

    float accAll = 0.f, accLo = 0.f;
    float capA = 0.f, capB = 0.f, capY0 = 0.f, capY199 = 0.f, capU = 0.f;

    for (int t = 0; t < 13; ++t) {
        const int m = 16 * t + p;
        const bool uSlot = (t == 12) && (p == 8);
        float g = uSlot ? u : (float)m * kStep;
        if (lane >= 16) g = 0.0f;             // upper lanes carry K 8..15, 24..31 -> zero

        v16h Bm = {};
        {
            _Float16 gh = (_Float16)g;
            _Float16 gl = (_Float16)(g - (float)gh);
            Bm[0] = gh; Bm[1] = gl; Bm[2] = gh;
            Bm[3] = fh; Bm[4] = fl; Bm[5] = fh;
        }

        // h_pre tiles: D[m][n] = spW1_g[m]*g_n + spW1_f[m]*fx + b1[m]
        v8f D0 = __builtin_amdgcn_wmma_f32_16x16x32_f16(
                     false, A0, false, Bm, (short)0, C0, false, false);
        v8f D1 = __builtin_amdgcn_wmma_f32_16x16x32_f16(
                     false, A1, false, Bm, (short)0, C1, false, false);

        // sigmoid + partial dot over this lane's 16 hidden values
        float pd = 0.f;
#pragma unroll
        for (int r = 0; r < 8; ++r) {
            pd = fmaf(sigmoid_fast(D0[r]), s2a[r], pd);
            pd = fmaf(sigmoid_fast(D1[r]), s2b[r], pd);
        }
        pd += __shfl_xor(pd, 16);             // complementary hidden rows
        const float y = softplus_fast(pd + bias2);

        // bookkeeping for trapezoid partial/total sums
        if (uSlot) capU = y;
        if (m < kND) {
            accAll += y;
            if (m >= 1 && m <= idx - 2) accLo += y;
            if (m == idx - 1)  capA   = y;
            if (m == idx)      capB   = y;
            if (m == 0)        capY0  = y;
            if (m == kND - 1)  capY199 = y;
        }
    }

    // butterfly reductions within 16-lane halves (halves are duplicates)
#pragma unroll
    for (int s = 1; s < 16; s <<= 1) {
        accAll += __shfl_xor(accAll, s);
        accLo  += __shfl_xor(accLo, s);
    }
    const float yA   = __shfl(capA, (idx - 1) & 15);
    const float yB   = __shfl(capB, idx & 15);
    const float Y0   = __shfl(capY0, 0);
    const float Y199 = __shfl(capY199, 7);     // m=199 -> tile 12, slot 7
    const float yu   = __shfl(capU, 8);

    // cumtrapz at insertion point / total, reconstructed analytically
    const float gA    = (float)(idx - 1) * kStep;
    const float gB    = (float)idx * kStep;
    const float Pprev = (idx >= 2) ? kStep * (0.5f * (Y0 + yA) + accLo) : 0.0f;
    const float Ttot  = kStep * (0.5f * (Y0 + Y199) + (accAll - Y0 - Y199));
    const float wa    = 0.5f * (u - gA) * (yA + yu);
    const float wb    = 0.5f * (gB - u) * (yu + yB);
    const float Sidx  = Pprev + wa;
    const float Stot  = Ttot - 0.5f * kStep * (yA + yB) + wa + wb;

    float val = Sidx / Stot;
    val = fminf(fmaxf(val, kEps), 1.0f - kEps);
    if (idx == 0) val = kEps;                  // u == 0 -> z = 0 -> clip

    if (lane == 0) out[task] = val;
}

extern "C" void kernel_launch(void* const* d_in, const int* in_sizes, int n_in,
                              void* d_out, int out_size, void* d_ws, size_t ws_size,
                              hipStream_t stream) {
    (void)in_sizes; (void)n_in; (void)d_ws; (void)ws_size; (void)out_size;
    const float* U  = (const float*)d_in[0];
    const float* W1 = (const float*)d_in[1];
    const float* b1 = (const float*)d_in[2];
    const float* W2 = (const float*)d_in[3];
    const float* b2 = (const float*)d_in[4];
    float* out = (float*)d_out;

    const int tasks = 2 * kB;                  // 32768 (sample, axis) tasks
    copula_marginal_kernel<<<tasks / 8, 256, 0, stream>>>(U, W1, b1, W2, b2, out);
}
